// MultiHeadAttention_69389491634882
// MI455X (gfx1250) — matrix operand order repaired and checked by reading, not compiled, not run on hardware
//
#include <hip/hip_runtime.h>
#include <hip/hip_bf16.h>

typedef __attribute__((ext_vector_type(16))) _Float16 v16h;
typedef __attribute__((ext_vector_type(8)))  _Float16 v8h;
typedef __attribute__((ext_vector_type(4)))  _Float16 v4h;
typedef __attribute__((ext_vector_type(8)))  float    v8f;
typedef int vi4 __attribute__((vector_size(16)));   // matches builtin param type

constexpr int Bb  = 4;
constexpr int Tt  = 2048;
constexpr int Ee  = 1024;
constexpr int Dd  = 64;
constexpr int NHh = 16;
constexpr int Rr  = Bb * Tt;     // 8192 total rows

// ---------------------------------------------------------------------------
// CDNA5 async global->LDS copy (ASYNCcnt-tracked). Guarded so the file always
// compiles; fallback is a synchronous LDS copy (DS in-order keeps it correct).
// ---------------------------------------------------------------------------
#if __has_builtin(__builtin_amdgcn_global_load_async_to_lds_b128)
#define HAVE_ASYNC_LDS 1
static __device__ inline void async_cp16(_Float16* lds, const _Float16* glb) {
  __builtin_amdgcn_global_load_async_to_lds_b128(
      (__attribute__((address_space(1))) vi4*)glb,
      (__attribute__((address_space(3))) vi4*)lds, 0, 0);
}
#else
#define HAVE_ASYNC_LDS 0
static __device__ inline void async_cp16(_Float16* lds, const _Float16* glb) {
  *(v8h*)lds = *(const v8h*)glb;   // sync fallback
}
#endif

#if __has_builtin(__builtin_amdgcn_s_wait_asynccnt)
#define WAIT_ASYNC(n) __builtin_amdgcn_s_wait_asynccnt(n)
#else
#define WAIT_ASYNC(n) asm volatile("s_wait_asynccnt %0" ::"i"(n) : "memory")
#endif

#if HAVE_ASYNC_LDS
#define WAIT_SYNC_FALLBACK()
#else
#define WAIT_SYNC_FALLBACK() asm volatile("s_wait_dscnt 0x0" ::: "memory")
#endif

static __device__ inline v8f vzero8f() {
  v8f r;
#pragma unroll
  for (int i = 0; i < 8; ++i) r[i] = 0.0f;
  return r;
}

static __device__ inline float halfmax16(float t) {
  t = fmaxf(t, __shfl_xor(t, 1, 32));
  t = fmaxf(t, __shfl_xor(t, 2, 32));
  t = fmaxf(t, __shfl_xor(t, 4, 32));
  t = fmaxf(t, __shfl_xor(t, 8, 32));
  return t;
}
static __device__ inline float halfsum16(float t) {
  t += __shfl_xor(t, 1, 32);
  t += __shfl_xor(t, 2, 32);
  t += __shfl_xor(t, 4, 32);
  t += __shfl_xor(t, 8, 32);
  return t;
}

// ---------------------------------------------------------------------------
// Kernel 1: convert activations f32 -> f16
// ---------------------------------------------------------------------------
__global__ __launch_bounds__(256)
void cvt_x_f16(const float* __restrict__ x, _Float16* __restrict__ xh) {
  size_t i = ((size_t)blockIdx.x * blockDim.x + threadIdx.x) * 4;
  float4 v = *(const float4*)(x + i);
  v4h o;
  o[0] = (_Float16)v.x; o[1] = (_Float16)v.y;
  o[2] = (_Float16)v.z; o[3] = (_Float16)v.w;
  *(v4h*)(xh + i) = o;
}

// ---------------------------------------------------------------------------
// Kernel 2: transpose+convert weights: Wt[n][k] = (f16) W[k][n]
// ---------------------------------------------------------------------------
__global__ __launch_bounds__(256)
void transpose_w_f16(const float* __restrict__ Wq, const float* __restrict__ Wk,
                     const float* __restrict__ Wv, const float* __restrict__ Wo,
                     _Float16* __restrict__ WtQ, _Float16* __restrict__ WtK,
                     _Float16* __restrict__ WtV, _Float16* __restrict__ WtO) {
  __shared__ float tile[32][33];
  int z = blockIdx.z;
  const float* W = (z == 0) ? Wq : (z == 1) ? Wk : (z == 2) ? Wv : Wo;
  _Float16*   Wt = (z == 0) ? WtQ : (z == 1) ? WtK : (z == 2) ? WtV : WtO;
  int tx = threadIdx.x;            // 0..31
  int ty = threadIdx.y;            // 0..7
  int k0 = blockIdx.x * 32;
  int n0 = blockIdx.y * 32;
#pragma unroll
  for (int i = 0; i < 4; ++i)
    tile[ty + 8 * i][tx] = W[(size_t)(k0 + ty + 8 * i) * Ee + n0 + tx];
  __syncthreads();
#pragma unroll
  for (int i = 0; i < 4; ++i)
    Wt[(size_t)(n0 + ty + 8 * i) * Ee + k0 + tx] = (_Float16)tile[tx][ty + 8 * i];
}

// ---------------------------------------------------------------------------
// Kernel 3: QKV projection GEMM, WMMA f32_16x16x32_f16 fed from async-staged
// LDS tiles (TRIPLE buffered: 2 tiles in flight).  Block = 256 thr (8 waves),
// tile 64(M)x128(N), K-step 32.  z=0: Q (scaled), z=1: K, z=2: V (transposed).
// ---------------------------------------------------------------------------
__global__ __launch_bounds__(256)
void qkv_gemm(const _Float16* __restrict__ xh,
              const _Float16* __restrict__ WtQ, const _Float16* __restrict__ WtK,
              const _Float16* __restrict__ WtV,
              const float* __restrict__ bq, const float* __restrict__ bk,
              const float* __restrict__ bv,
              _Float16* __restrict__ Qh, _Float16* __restrict__ Kh,
              _Float16* __restrict__ Vt) {
  __shared__ __align__(32) _Float16 sA[3][64][32];    // 12 KB
  __shared__ __align__(32) _Float16 sB[3][128][32];   // 24 KB

  int z = blockIdx.z;
  const _Float16* Wt  = (z == 0) ? WtQ : (z == 1) ? WtK : WtV;
  const float*    bia = (z == 0) ? bq  : (z == 1) ? bk  : bv;

  int tid  = threadIdx.x;
  int lane = tid & 31;
  int w    = tid >> 5;          // 0..7
  int wr   = w & 3;             // wave row
  int wc   = w >> 2;            // wave col
  int c    = lane & 15;
  int half = lane >> 4;
  int m0blk = blockIdx.x * 64;
  int n0blk = blockIdx.y * 128;
  int m0 = m0blk + wr * 16;
  int n0 = n0blk + wc * 64;

  // per-thread async copy slots: A tile 64x32, B tile 128x32 (f16)
  int row = tid >> 2;           // 0..63
  int seg = tid & 3;            // 16-byte segment

  auto issue_tile = [&](int buf, int k) {
    async_cp16(&sA[buf][row][seg * 8],
               xh + (size_t)(m0blk + row) * Ee + k + seg * 8);
    async_cp16(&sB[buf][row][seg * 8],
               Wt + (size_t)(n0blk + row) * Ee + k + seg * 8);
    async_cp16(&sB[buf][row + 64][seg * 8],
               Wt + (size_t)(n0blk + row + 64) * Ee + k + seg * 8);
  };

  v8f acc[4];
#pragma unroll
  for (int j = 0; j < 4; ++j) acc[j] = vzero8f();

  issue_tile(0, 0);
  issue_tile(1, 32);
  int it = 0;
  for (int k = 0; k < Ee; k += 32, ++it) {
    int cur = it % 3;
    // tile `it` resident once the newest batch (3 copies) is the only one left
    if (k + 32 < Ee) { WAIT_ASYNC(3); } else { WAIT_ASYNC(0); }
    WAIT_SYNC_FALLBACK();
    __syncthreads();             // publish to all waves
    if (k + 64 < Ee) issue_tile((it + 2) % 3, k + 64);  // 2-deep prefetch

    // Batch all fragment loads, then issue the 4 WMMAs back-to-back.
    v16h a;
    const _Float16* aL = &sA[cur][wr * 16 + c][half * 16];
    ((v8h*)&a)[0] = *(const v8h*)aL;
    ((v8h*)&a)[1] = *(const v8h*)(aL + 8);
    v16h bfr[4];
#pragma unroll
    for (int j = 0; j < 4; ++j)
      bfr[j] = *(const v16h*)&sB[cur][wc * 64 + 16 * j + c][half * 16];
#pragma unroll
    for (int j = 0; j < 4; ++j)
      acc[j] = __builtin_amdgcn_wmma_f32_16x16x32_f16(
          false, a, false, bfr[j], (short)0, acc[j], false, false);
  }

  // Epilogue: bias (+Q scale), convert, scatter to head-split layouts.
#pragma unroll
  for (int j = 0; j < 4; ++j) {
    int n = n0 + 16 * j + c;           // output feature
    float bval = bia[n];
    int h = n >> 6;                     // head
    int d = n & 63;                     // dim within head
    if (z < 2) {
      _Float16* outp = (z == 0) ? Qh : Kh;
      float scale = (z == 0) ? 0.125f : 1.0f;
#pragma unroll
      for (int v = 0; v < 8; ++v) {
        int rr = m0 + v + 8 * half;
        int bbx = rr >> 11;
        int t   = rr & (Tt - 1);
        outp[(((size_t)(bbx * NHh + h) * Tt + t) * Dd) + d] =
            (_Float16)((acc[j][v] + bval) * scale);
      }
    } else {
      v8h pk;
#pragma unroll
      for (int v = 0; v < 8; ++v) pk[v] = (_Float16)(acc[j][v] + bval);
      int rr0 = m0 + 8 * half;
      int bbx = rr0 >> 11;
      int t0  = rr0 & (Tt - 1);
      *(v8h*)(Vt + ((size_t)(bbx * NHh + h) * Dd + d) * Tt + t0) = pk;
    }
  }
}

// ---------------------------------------------------------------------------
// Kernel 4: causal flash attention.  One wave per (b,h,16-query tile).
// K/V chunks async-staged to LDS, TRIPLE buffered (2 chunks in flight;
// wave-private so no barriers, only s_wait_asynccnt thresholds).
// Per 32-key chunk: 4 WMMA QK^T, online softmax, P reshape via LDS, 4 WMMA PV.
// ---------------------------------------------------------------------------
__global__ __launch_bounds__(32)
void flash_attn(const _Float16* __restrict__ Qh, const _Float16* __restrict__ Kh,
                const _Float16* __restrict__ Vt, _Float16* __restrict__ Ah) {
  __shared__ __align__(32) _Float16 sK[3][32][64];   // 12 KB
  __shared__ __align__(32) _Float16 sV[3][64][32];   // 12 KB
  __shared__ __align__(32) _Float16 sP[16 * 32];     // 1 KB

  int lane = threadIdx.x & 31;
  int c    = lane & 15;
  int half = lane >> 4;
  int qt0  = blockIdx.x * 16;
  int bh   = blockIdx.y;                 // b*NH + h
  const _Float16* Qb = Qh + (size_t)bh * Tt * Dd;
  const _Float16* Kb = Kh + (size_t)bh * Tt * Dd;
  const _Float16* Vb = Vt + (size_t)bh * Dd * Tt;

  auto issue_kv = [&](int buf, int key0) {
#pragma unroll
    for (int u = 0; u < 8; ++u)          // K: 32 rows x 128 B, lane owns a row
      async_cp16(&sK[buf][lane][u * 8],
                 Kb + (size_t)(key0 + lane) * Dd + u * 8);
#pragma unroll
    for (int i = 0; i < 2; ++i)          // V: 64 rows x 64 B, lane owns 2 rows
#pragma unroll
      for (int s4 = 0; s4 < 4; ++s4)
        async_cp16(&sV[buf][lane + 32 * i][s4 * 8],
                   Vb + (size_t)(lane + 32 * i) * Tt + key0 + s4 * 8);
  };

  // Q A-fragments for the two 32-wide K-dim steps (D = 64), loop-resident.
  v16h aq[2];
#pragma unroll
  for (int s = 0; s < 2; ++s) {
    const _Float16* p = Qb + (size_t)(qt0 + c) * Dd + 32 * s + half * 16;
    ((v8h*)&aq[s])[0] = *(const v8h*)p;
    ((v8h*)&aq[s])[1] = *(const v8h*)(p + 8);
  }

  float mrow[8], lrow[8];
#pragma unroll
  for (int v = 0; v < 8; ++v) { mrow[v] = -__builtin_inff(); lrow[v] = 0.0f; }
  v8f o[4];
#pragma unroll
  for (int j = 0; j < 4; ++j) o[j] = vzero8f();

  int nchunk = (qt0 >> 5) + 1;           // causal: chunks of 32 keys
  issue_kv(0, 0);
  if (nchunk > 1) issue_kv(1, 32);
  for (int ch = 0; ch < nchunk; ++ch) {
    int key0 = ch * 32;
    int cur  = ch % 3;
    // 2-deep prefetch; wait so chunk `ch` (16 copies) is resident.
    if (ch + 2 < nchunk) {
      issue_kv((ch + 2) % 3, key0 + 64);
      WAIT_ASYNC(32);
    } else if (ch + 1 < nchunk) {
      WAIT_ASYNC(16);
    } else {
      WAIT_ASYNC(0);
    }
    WAIT_SYNC_FALLBACK();

    // ---- S = Q K^T (scale pre-folded into Q); batch loads then WMMAs ----
    v16h bk[4];
#pragma unroll
    for (int j = 0; j < 2; ++j)
#pragma unroll
      for (int s = 0; s < 2; ++s)
        bk[2 * j + s] = *(const v16h*)&sK[cur][16 * j + c][32 * s + half * 16];
    v8f S[2];
#pragma unroll
    for (int j = 0; j < 2; ++j) S[j] = vzero8f();
#pragma unroll
    for (int j = 0; j < 2; ++j)
#pragma unroll
      for (int s = 0; s < 2; ++s)
        S[j] = __builtin_amdgcn_wmma_f32_16x16x32_f16(
            false, aq[s], false, bk[2 * j + s], (short)0, S[j], false, false);

    // ---- causal mask ----
#pragma unroll
    for (int j = 0; j < 2; ++j) {
      int kn = key0 + 16 * j + c;
#pragma unroll
      for (int v = 0; v < 8; ++v) {
        int qr = qt0 + v + 8 * half;
        if (kn > qr) S[j][v] = -__builtin_inff();
      }
    }

    // ---- online softmax (row stats across 16-lane halves) ----
    float mnew[8], fscale[8];
#pragma unroll
    for (int v = 0; v < 8; ++v) {
      float t = fmaxf(S[0][v], S[1][v]);
      t = halfmax16(t);
      mnew[v] = fmaxf(mrow[v], t);
    }
#pragma unroll
    for (int j = 0; j < 2; ++j)
#pragma unroll
      for (int v = 0; v < 8; ++v) S[j][v] = expf(S[j][v] - mnew[v]);
#pragma unroll
    for (int v = 0; v < 8; ++v) {
      float rs = halfsum16(S[0][v] + S[1][v]);
      fscale[v] = expf(mrow[v] - mnew[v]);   // 0 when mrow == -inf
      lrow[v]   = lrow[v] * fscale[v] + rs;
      mrow[v]   = mnew[v];
    }
#pragma unroll
    for (int j = 0; j < 4; ++j)
#pragma unroll
      for (int v = 0; v < 8; ++v) o[j][v] *= fscale[v];

    // ---- reshape P (C-layout) -> A-fragment via LDS ----
#pragma unroll
    for (int j = 0; j < 2; ++j)
#pragma unroll
      for (int v = 0; v < 8; ++v)
        sP[(v + 8 * half) * 32 + 16 * j + c] = (_Float16)S[j][v];
    asm volatile("s_wait_dscnt 0x0" ::: "memory");  // LDS in-order per wave
    v16h ap;
    ((v8h*)&ap)[0] = *(const v8h*)(sP + c * 32 + half * 16);
    ((v8h*)&ap)[1] = *(const v8h*)(sP + c * 32 + 8 + half * 16);

    // ---- O += P V ; batch loads then WMMAs ----
    v16h bv[4];
#pragma unroll
    for (int j2 = 0; j2 < 4; ++j2)
      bv[j2] = *(const v16h*)&sV[cur][16 * j2 + c][half * 16];
#pragma unroll
    for (int j2 = 0; j2 < 4; ++j2)
      o[j2] = __builtin_amdgcn_wmma_f32_16x16x32_f16(
          false, ap, false, bv[j2], (short)0, o[j2], false, false);
  }

  // ---- normalize and store attn output in [B,T,E] (heads merged) ----
  int bbx = bh >> 4;
  int h   = bh & 15;
#pragma unroll
  for (int j2 = 0; j2 < 4; ++j2) {
#pragma unroll
    for (int v = 0; v < 8; ++v) {
      int t = qt0 + v + 8 * half;
      Ah[((size_t)(bbx * Tt + t)) * Ee + h * 64 + 16 * j2 + c] =
          (_Float16)(o[j2][v] / lrow[v]);
    }
  }
}

// ---------------------------------------------------------------------------
// Kernel 5: output projection GEMM (same triple-buffered async WMMA pipeline,
// f32 epilogue straight to d_out)
// ---------------------------------------------------------------------------
__global__ __launch_bounds__(256)
void out_gemm(const _Float16* __restrict__ Ah, const _Float16* __restrict__ WtO,
              const float* __restrict__ bo, float* __restrict__ out) {
  __shared__ __align__(32) _Float16 sA[3][64][32];
  __shared__ __align__(32) _Float16 sB[3][128][32];

  int tid  = threadIdx.x;
  int lane = tid & 31;
  int w    = tid >> 5;
  int wr   = w & 3;
  int wc   = w >> 2;
  int c    = lane & 15;
  int half = lane >> 4;
  int m0blk = blockIdx.x * 64;
  int n0blk = blockIdx.y * 128;
  int m0 = m0blk + wr * 16;
  int n0 = n0blk + wc * 64;

  int row = tid >> 2;
  int seg = tid & 3;

  auto issue_tile = [&](int buf, int k) {
    async_cp16(&sA[buf][row][seg * 8],
               Ah + (size_t)(m0blk + row) * Ee + k + seg * 8);
    async_cp16(&sB[buf][row][seg * 8],
               WtO + (size_t)(n0blk + row) * Ee + k + seg * 8);
    async_cp16(&sB[buf][row + 64][seg * 8],
               WtO + (size_t)(n0blk + row + 64) * Ee + k + seg * 8);
  };

  v8f acc[4];
#pragma unroll
  for (int j = 0; j < 4; ++j) acc[j] = vzero8f();

  issue_tile(0, 0);
  issue_tile(1, 32);
  int it = 0;
  for (int k = 0; k < Ee; k += 32, ++it) {
    int cur = it % 3;
    if (k + 32 < Ee) { WAIT_ASYNC(3); } else { WAIT_ASYNC(0); }
    WAIT_SYNC_FALLBACK();
    __syncthreads();
    if (k + 64 < Ee) issue_tile((it + 2) % 3, k + 64);

    v16h a;
    const _Float16* aL = &sA[cur][wr * 16 + c][half * 16];
    ((v8h*)&a)[0] = *(const v8h*)aL;
    ((v8h*)&a)[1] = *(const v8h*)(aL + 8);
    v16h bfr[4];
#pragma unroll
    for (int j = 0; j < 4; ++j)
      bfr[j] = *(const v16h*)&sB[cur][wc * 64 + 16 * j + c][half * 16];
#pragma unroll
    for (int j = 0; j < 4; ++j)
      acc[j] = __builtin_amdgcn_wmma_f32_16x16x32_f16(
          false, a, false, bfr[j], (short)0, acc[j], false, false);
  }
#pragma unroll
  for (int j = 0; j < 4; ++j) {
    int n = n0 + 16 * j + c;
    float bval = bo[n];
#pragma unroll
    for (int v = 0; v < 8; ++v) {
      int rr = m0 + v + 8 * half;
      out[(size_t)rr * Ee + n] = acc[j][v] + bval;
    }
  }
}

// ---------------------------------------------------------------------------
extern "C" void kernel_launch(void* const* d_in, const int* in_sizes, int n_in,
                              void* d_out, int out_size, void* d_ws, size_t ws_size,
                              hipStream_t stream) {
  const float* x  = (const float*)d_in[0];
  const float* Wq = (const float*)d_in[1];
  const float* bq = (const float*)d_in[2];
  const float* Wk = (const float*)d_in[3];
  const float* bk = (const float*)d_in[4];
  const float* Wv = (const float*)d_in[5];
  const float* bv = (const float*)d_in[6];
  const float* Wo = (const float*)d_in[7];
  const float* bo = (const float*)d_in[8];
  float* out = (float*)d_out;

  char* ws = (char*)d_ws;
  const size_t MB = (size_t)1 << 20;
  _Float16* xh  = (_Float16*)(ws + 0 * MB);    // 16 MB  [R,E]
  _Float16* WtQ = (_Float16*)(ws + 16 * MB);   // 2 MB each, N-major [E][E]
  _Float16* WtK = WtQ + (size_t)Ee * Ee;
  _Float16* WtV = WtK + (size_t)Ee * Ee;
  _Float16* WtO = WtV + (size_t)Ee * Ee;
  _Float16* Qh  = (_Float16*)(ws + 24 * MB);   // 16 MB  [b,h,t,d]
  _Float16* Kh  = (_Float16*)(ws + 40 * MB);   // 16 MB  [b,h,t,d]
  _Float16* Vt  = (_Float16*)(ws + 56 * MB);   // 16 MB  [b,h,d,t]
  _Float16* Ah  = (_Float16*)(ws + 72 * MB);   // 16 MB  [B,T,E]

  cvt_x_f16<<<(Rr * Ee) / (256 * 4), 256, 0, stream>>>(x, xh);
  transpose_w_f16<<<dim3(Ee / 32, Ee / 32, 4), dim3(32, 8), 0, stream>>>(
      Wq, Wk, Wv, Wo, WtQ, WtK, WtV, WtO);
  qkv_gemm<<<dim3(Rr / 64, Ee / 128, 3), 256, 0, stream>>>(
      xh, WtQ, WtK, WtV, bq, bk, bv, Qh, Kh, Vt);
  flash_attn<<<dim3(Tt / 16, Bb * NHh), 32, 0, stream>>>(Qh, Kh, Vt, Ah);
  out_gemm<<<dim3(Rr / 64, Ee / 128), 256, 0, stream>>>(Ah, WtO, bo, out);
}
